// AbLangGAT_69492570849815
// MI455X (gfx1250) — compile-verified, matrix operand order checked by reading, not run on hardware
//
#include <hip/hip_runtime.h>
#include <hip/hip_bf16.h>

// ---------------------------------------------------------------------------
// Model constants (from reference)
// ---------------------------------------------------------------------------
#define NN      40000
#define EE      480000
#define ETOT    520000      // E + N self loops
#define BATCH   160
#define NPG     250
#define HEADS   4
#define DCAT    3072
#define HID     1536

typedef __attribute__((ext_vector_type(16))) __bf16 v16bf;
typedef __attribute__((ext_vector_type(8)))  float  v8f;
// GCC-style vectors to match amdgcn builtin prototypes exactly
typedef int          gv4i __attribute__((vector_size(16)));
typedef int          gv8i __attribute__((vector_size(32)));
typedef unsigned int gv4u __attribute__((vector_size(16)));

// --------------------- CDNA5 feature detection (device pass only) ----------
#ifndef __has_builtin
#define __has_builtin(x) 0
#endif
#if defined(__AMDGCN__) && __has_builtin(__builtin_amdgcn_global_load_async_to_lds_b128)
#define HAVE_ASYNC_LDS 1
#else
#define HAVE_ASYNC_LDS 0
#endif
#if defined(__AMDGCN__) && __has_builtin(__builtin_amdgcn_s_wait_asynccnt)
#define HAVE_WAIT_ASYNC 1
#else
#define HAVE_WAIT_ASYNC 0
#endif
#if defined(__AMDGCN__) && __has_builtin(__builtin_amdgcn_tensor_load_to_lds)
#define HAVE_TDM 1
#else
#define HAVE_TDM 0
#endif
#if defined(__AMDGCN__) && __has_builtin(__builtin_amdgcn_s_wait_tensorcnt)
#define HAVE_WAIT_TENSOR 1
#else
#define HAVE_WAIT_TENSOR 0
#endif

#if defined(__AMDGCN__)
typedef __attribute__((address_space(1))) void* as1vp;
typedef __attribute__((address_space(3))) void* as3vp;
typedef __attribute__((address_space(3))) char* as3cp;
typedef __attribute__((address_space(1))) gv4i* as1_v4ip;
typedef __attribute__((address_space(3))) gv4i* as3_v4ip;
#endif

__device__ __forceinline__ void wait_async0() {
#if HAVE_WAIT_ASYNC
    __builtin_amdgcn_s_wait_asynccnt(0);
#elif HAVE_ASYNC_LDS
    asm volatile("s_wait_asynccnt 0x0" ::: "memory");
#endif
}

__device__ __forceinline__ void wait_tensor0() {
#if HAVE_WAIT_TENSOR
    __builtin_amdgcn_s_wait_tensorcnt(0);
#elif HAVE_TDM
    asm volatile("s_wait_tensorcnt 0x0" ::: "memory");
#endif
}

// ---------------------------------------------------------------------------
// fp32 -> bf16 conversion (strided source, dense dest)
// ---------------------------------------------------------------------------
__global__ void conv_bf16_kernel(const float* __restrict__ src, int srcLd, int cols,
                                 __hip_bfloat16* __restrict__ dst, size_t total) {
    size_t gid = (size_t)blockIdx.x * blockDim.x + threadIdx.x;
    if (gid >= total) return;
    size_t r = gid / (size_t)cols;
    int    c = (int)(gid - r * (size_t)cols);
    dst[gid] = __float2bfloat16(src[r * (size_t)srcLd + c]);
}

// ---------------------------------------------------------------------------
// Pack row-major K x Nc fp32 weights into WMMA B-fragment order (bf16):
//   Bp[((kt*ntiles + nt)*32 + lane)*16 + elem]
//   lane: col = nt*16 + (lane&15), kbase = (lane>=16)?8:0
//   elem<8 : K = kt*32 + kbase + elem ; elem>=8 : K = kt*32 + 16 + kbase + elem-8
// ---------------------------------------------------------------------------
__global__ void pack_B_kernel(const float* __restrict__ W, __hip_bfloat16* __restrict__ Bp,
                              int K, int Nc) {
    size_t gid = (size_t)blockIdx.x * blockDim.x + threadIdx.x;
    size_t total = (size_t)K * (size_t)Nc;
    if (gid >= total) return;
    int elem = (int)(gid & 15);
    size_t t = gid >> 4;
    int lane = (int)(t & 31); t >>= 5;
    int ntiles = Nc >> 4;
    int nt = (int)(t % (size_t)ntiles);
    int kt = (int)(t / (size_t)ntiles);
    int col = nt * 16 + (lane & 15);
    int kbase = (lane >> 4) << 3;
    int k = (elem < 8) ? (kt * 32 + kbase + elem)
                       : (kt * 32 + 16 + kbase + (elem - 8));
    Bp[gid] = __float2bfloat16(W[(size_t)k * Nc + col]);
}

// ---------------------------------------------------------------------------
// Tiled bf16 WMMA GEMM: C[M,Nc] = A[M,K] * B[K,Nc] (+bias), fp32 accumulate.
// Block = 256 threads = 8 waves. Block tile 128(M) x 64(N); wave tile 16x64.
//
// Double-buffered LDS pipeline per 32-wide K step:
//   * B panel (4KB, pre-packed fragment order) staged with
//     global_load_async_to_lds_b128 (ASYNCcnt) by all 256 threads
//   * A panel (128x32 bf16, row stride K) staged by the Tensor Data Mover
//     (tensor_load_to_lds, TENSORcnt), issued by wave 0; tensor_dim1 = M so
//     the ragged final M-block is zero-filled by TDM OOB semantics
// Copies for step kt+1 overlap the 4 WMMAs of step kt. All fragment LDS
// loads are issued before the WMMA group so the ds clause drains with a
// single s_wait_dscnt and the 4 WMMAs issue back-to-back.
// ---------------------------------------------------------------------------
__global__ __launch_bounds__(256) void wmma_gemm_bf16(
    const __hip_bfloat16* __restrict__ A, const __hip_bfloat16* __restrict__ Bp,
    float* __restrict__ C, const float* __restrict__ bias,
    int M, int K, int Nc)
{
    __shared__ __align__(16) char sB[2][4096];
#if HAVE_TDM
    __shared__ __align__(16) char sA[2][8192];
#endif

    const int tid  = threadIdx.x;
    const int lane = tid & 31;
    const int wave = tid >> 5;
    const int m0   = blockIdx.y * 128 + wave * 16;
    const int n0   = blockIdx.x * 64;
    const int ntiles = Nc >> 4;
    const int ktiles = K >> 5;
    const int kbase  = (lane >> 4) << 3;    // 0 or 8
    const int rlo    = lane & 15;

    // ---- B panel staging: 256 threads x 16B = 4KB (contiguous in packed Bp)
    auto stageB = [&](int kt, int buf) {
        const char* src = (const char*)(Bp + ((size_t)kt * ntiles + (n0 >> 4)) * 512) + tid * 16;
        char* dst = &sB[buf][tid * 16];
#if HAVE_ASYNC_LDS
        __builtin_amdgcn_global_load_async_to_lds_b128(
            (as1_v4ip)(as1vp)(void*)src, (as3_v4ip)(as3vp)dst, 0, 0);
#else
        *(uint4*)dst = *(const uint4*)src;
#endif
    };

#if HAVE_TDM
    // ---- A panel staging via TDM (issued by wave 0 only; uniform operands)
    auto stageA = [&](int kt, int buf) {
        unsigned ldsOff = (unsigned)(unsigned long long)(as3cp)&sA[buf][0];
        unsigned long long ga = (unsigned long long)
            (A + (size_t)blockIdx.y * 128 * (size_t)K + (size_t)kt * 32);
        gv4u g0;
        g0[0] = 1u;                                        // count=1 (valid user D#)
        g0[1] = ldsOff;                                    // lds_addr
        g0[2] = (unsigned)(ga & 0xFFFFFFFFull);            // global_addr[31:0]
        g0[3] = (unsigned)((ga >> 32) & 0x1FFFFFFull)      // global_addr[56:32]
              | (2u << 30);                                // type=2 ("image")
        unsigned long long st0 = (unsigned long long)K;    // dim0 stride (elements)
        gv8i g1;
        g1[0] = 0x10000;                                   // data_size=1 -> 2 bytes
        g1[1] = (int)(((unsigned)K & 0xFFFFu) << 16);      // tensor_dim0[15:0]
        g1[2] = (int)(((unsigned)K >> 16) | (((unsigned)M & 0xFFFFu) << 16));
        g1[3] = (int)(((unsigned)M >> 16) | (32u << 16));  // tile_dim0 = 32
        g1[4] = 128;                                       // tile_dim1=128, tile_dim2=0
        g1[5] = (int)(st0 & 0xFFFFFFFFull);                // dim0_stride[31:0]
        g1[6] = (int)((st0 >> 32) & 0xFFFFull);            // dim0_stride[47:32]
        g1[7] = 0;
        gv4i z4 = {0, 0, 0, 0};
        gv8i z8 = {0, 0, 0, 0, 0, 0, 0, 0};
        __builtin_amdgcn_tensor_load_to_lds(g0, g1, z4, z4, z8, 0);
    };
#else
    int arow = m0 + rlo;
    if (arow >= M) arow = M - 1;                           // clamp loads, guard stores
    const __hip_bfloat16* aptr = A + (size_t)arow * (size_t)K + kbase;
#endif

    // ---- prologue: stage step 0
#if HAVE_TDM
    if (tid < 32) { stageA(0, 0); wait_tensor0(); }
#endif
    stageB(0, 0);
    wait_async0();
    __syncthreads();

    v8f acc[4] = {};

    for (int kt = 0; kt < ktiles; ++kt) {
        const int cur = kt & 1;
        // kick off copies for next step (overlaps with WMMA below)
        if (kt + 1 < ktiles) {
#if HAVE_TDM
            if (tid < 32) stageA(kt + 1, cur ^ 1);
#endif
            stageB(kt + 1, cur ^ 1);
        }

        // A fragment: K = kbase + {0..7}, 16 + kbase + {0..7}
        v16bf afrag;
#if HAVE_TDM
        {
            const char* ab = &sA[cur][(wave * 16 + rlo) * 64 + kbase * 2];
            reinterpret_cast<uint4*>(&afrag)[0] = *(const uint4*)(ab);
            reinterpret_cast<uint4*>(&afrag)[1] = *(const uint4*)(ab + 32);
        }
#else
        {
            const uint4* ap = reinterpret_cast<const uint4*>(aptr + kt * 32);
            reinterpret_cast<uint4*>(&afrag)[0] = ap[0];
            reinterpret_cast<uint4*>(&afrag)[1] = ap[2];
        }
#endif

        // issue all B-fragment LDS loads first (one ds clause, one wait) ...
        v16bf bfrag[4];
#pragma unroll
        for (int j = 0; j < 4; ++j) {
            const uint4* bp = reinterpret_cast<const uint4*>(&sB[cur][(j * 32 + lane) * 32]);
            reinterpret_cast<uint4*>(&bfrag[j])[0] = bp[0];
            reinterpret_cast<uint4*>(&bfrag[j])[1] = bp[1];
        }
        // ... then the 4 WMMAs issue back-to-back on the matrix pipe
#pragma unroll
        for (int j = 0; j < 4; ++j) {
            acc[j] = __builtin_amdgcn_wmma_f32_16x16x32_bf16(
                false, afrag, false, bfrag[j], (short)0, acc[j], false, false);
        }

        // drain next-step copies, then publish buffers
        if (kt + 1 < ktiles) {
            wait_async0();
#if HAVE_TDM
            if (tid < 32) wait_tensor0();
#endif
        }
        __syncthreads();
    }

    // C/D layout: VGPR r -> row = m0 + r + ((lane>=16)?8:0) ; col = n0 + j*16 + (lane&15)
    const int rOff = (lane >> 4) << 3;
#pragma unroll
    for (int j = 0; j < 4; ++j) {
        int col = n0 + j * 16 + rlo;
        float badd = bias ? bias[col] : 0.0f;
#pragma unroll
        for (int r = 0; r < 8; ++r) {
            int row = m0 + r + rOff;
            if (row < M) C[(size_t)row * Nc + col] = acc[j][r] + badd;
        }
    }
}

// ---------------------------------------------------------------------------
// Per-node attention scores: s_src[n,h] = sum_c h[n,h,c]*a_src[h,c]  (and dst)
// One wave32 per (node, head); shuffle reduce.
// ---------------------------------------------------------------------------
__global__ __launch_bounds__(128) void node_scores_kernel(
    const float* __restrict__ h, const float* __restrict__ asrc,
    const float* __restrict__ adst, float* __restrict__ ssrc,
    float* __restrict__ sdst, int Nn, int H, int Cc)
{
    int wid  = blockIdx.x * (blockDim.x >> 5) + (threadIdx.x >> 5);
    int lane = threadIdx.x & 31;
    if (wid >= Nn * H) return;
    int n = wid / H, hh = wid - n * H;
    const float* hp = h + (size_t)n * (size_t)(H * Cc) + (size_t)hh * Cc;
    float s1 = 0.f, s2 = 0.f;
    for (int c = lane; c < Cc; c += 32) {
        float v = hp[c];
        s1 += v * asrc[hh * Cc + c];
        s2 += v * adst[hh * Cc + c];
    }
#pragma unroll
    for (int off = 16; off; off >>= 1) {
        s1 += __shfl_down(s1, off, 32);
        s2 += __shfl_down(s2, off, 32);
    }
    if (lane == 0) { ssrc[wid] = s1; sdst[wid] = s2; }
}

// ---------------------------------------------------------------------------
// Init per-layer state: out tile = bias broadcast; m = -inf; denom = 0
// ---------------------------------------------------------------------------
__global__ void init_layer_kernel(float* __restrict__ out, const float* __restrict__ bias,
                                  int ld, int cols, int Nn,
                                  float* __restrict__ m, float* __restrict__ denom, int NH) {
    size_t gid = (size_t)blockIdx.x * blockDim.x + threadIdx.x;
    size_t total = (size_t)Nn * (size_t)cols;
    if (gid < total) {
        size_t r = gid / (size_t)cols;
        int    c = (int)(gid - r * (size_t)cols);
        out[r * (size_t)ld + c] = bias[c];
    }
    if (gid < (size_t)NH) { m[gid] = -1e30f; denom[gid] = 0.0f; }
}

__device__ __forceinline__ void atomicMaxFloat(float* addr, float value) {
    if (value >= 0.0f) atomicMax((int*)addr, __float_as_int(value));
    else               atomicMin((unsigned int*)addr, __float_as_uint(value));
}

// edge src/dst with implicit self loops appended
__device__ __forceinline__ void edge_sd(const int* ei, int e, int E, int& s, int& d) {
    if (e < E) { s = ei[e]; d = ei[E + e]; } else { s = d = e - E; }
}

// ---------------------------------------------------------------------------
// Pass 1: e = leaky_relu(s_src[src]+s_dst[dst]); segment max over dst (atomic)
// ---------------------------------------------------------------------------
__global__ void edge_logits_kernel(const int* __restrict__ ei,
                                   const float* __restrict__ ssrc, const float* __restrict__ sdst,
                                   float* __restrict__ ew, float* __restrict__ m,
                                   int E, int Etot, int H) {
    int gid = blockIdx.x * blockDim.x + threadIdx.x;
    if (gid >= Etot * H) return;
    int e = gid / H, hh = gid - e * H;
    int s, d; edge_sd(ei, e, E, s, d);
    float v = ssrc[s * H + hh] + sdst[d * H + hh];
    v = (v > 0.0f) ? v : 0.2f * v;                  // leaky_relu 0.2
    ew[gid] = v;
    atomicMaxFloat(&m[d * H + hh], v);
}

// ---------------------------------------------------------------------------
// Pass 2: w = exp(e - m[dst]); denom[dst] += w
// ---------------------------------------------------------------------------
__global__ void edge_exp_kernel(const int* __restrict__ ei, float* __restrict__ ew,
                                const float* __restrict__ m, float* __restrict__ denom,
                                int E, int Etot, int H) {
    int gid = blockIdx.x * blockDim.x + threadIdx.x;
    if (gid >= Etot * H) return;
    int e = gid / H, hh = gid - e * H;
    int s, d; edge_sd(ei, e, E, s, d);
    float w = expf(ew[gid] - m[d * H + hh]);
    ew[gid] = w;
    atomicAdd(&denom[d * H + hh], w);
}

// ---------------------------------------------------------------------------
// Pass 3: out[dst] += h[src] * alpha   (alpha = w / (denom+1e-16)); one block/edge
// ---------------------------------------------------------------------------
__global__ __launch_bounds__(256) void edge_agg_kernel(
    const int* __restrict__ ei, const float* __restrict__ h,
    const float* __restrict__ ew, const float* __restrict__ denom,
    float* __restrict__ out, int E, int H, int Cc, int ldOut) {
    int e = blockIdx.x;
    int s, d; edge_sd(ei, e, E, s, d);
    int HC = H * Cc;
    const float* hs = h + (size_t)s * (size_t)HC;
    float* op = out + (size_t)d * (size_t)ldOut;
    __shared__ float alpha[8];
    if (threadIdx.x < H)
        alpha[threadIdx.x] = ew[e * H + threadIdx.x] /
                             (denom[d * H + threadIdx.x] + 1e-16f);
    if (threadIdx.x == 0) __builtin_prefetch(hs, 0, 0);
    __syncthreads();
    for (int i = threadIdx.x; i < HC; i += 256) {
        int hh = i / Cc;
        atomicAdd(&op[i], hs[i] * alpha[hh]);
    }
}

// ---------------------------------------------------------------------------
// ReLU + mean pooling: batch_ids == n/250 exactly (contiguous graphs)
// ---------------------------------------------------------------------------
__global__ __launch_bounds__(256) void pool_mean_relu_kernel(
    const float* __restrict__ cbuf, float* __restrict__ pooled) {
    int d = blockIdx.x * blockDim.x + threadIdx.x;
    int b = blockIdx.y;
    if (d >= DCAT) return;
    const float* p = cbuf + (size_t)b * NPG * DCAT + d;
    float s = 0.f;
    for (int k = 0; k < NPG; ++k) {
        float v = p[(size_t)k * DCAT];
        s += (v > 0.f) ? v : 0.f;
    }
    pooled[(size_t)b * DCAT + d] = s * (1.0f / (float)NPG);
}

// ---------------------------------------------------------------------------
// ReLU then LayerNorm in place (one block per row)
// ---------------------------------------------------------------------------
__global__ __launch_bounds__(256) void relu_ln_kernel(float* __restrict__ h1,
                                                      const float* __restrict__ g,
                                                      const float* __restrict__ bln, int D) {
    int b = blockIdx.x;
    float* row = h1 + (size_t)b * D;
    __shared__ float red[256];
    float s = 0.f;
    for (int i = threadIdx.x; i < D; i += 256) {
        float v = row[i]; v = (v > 0.f) ? v : 0.f;
        row[i] = v; s += v;
    }
    red[threadIdx.x] = s; __syncthreads();
    for (int off = 128; off; off >>= 1) {
        if (threadIdx.x < off) red[threadIdx.x] += red[threadIdx.x + off];
        __syncthreads();
    }
    float mu = red[0] / (float)D; __syncthreads();
    float s2 = 0.f;
    for (int i = threadIdx.x; i < D; i += 256) { float dv = row[i] - mu; s2 += dv * dv; }
    red[threadIdx.x] = s2; __syncthreads();
    for (int off = 128; off; off >>= 1) {
        if (threadIdx.x < off) red[threadIdx.x] += red[threadIdx.x + off];
        __syncthreads();
    }
    float inv = rsqrtf(red[0] / (float)D + 1e-5f);
    __syncthreads();
    for (int i = threadIdx.x; i < D; i += 256)
        row[i] = (row[i] - mu) * inv * g[i] + bln[i];
}

// ---------------------------------------------------------------------------
// Final: logits[b] = dot(h1[b], Wd2) + bd2 + 0.5
// ---------------------------------------------------------------------------
__global__ __launch_bounds__(256) void final_dot_kernel(const float* __restrict__ h1,
                                                        const float* __restrict__ w,
                                                        const float* __restrict__ bd2,
                                                        float* __restrict__ out, int D) {
    int b = blockIdx.x;
    __shared__ float red[256];
    float s = 0.f;
    for (int i = threadIdx.x; i < D; i += 256) s += h1[(size_t)b * D + i] * w[i];
    red[threadIdx.x] = s; __syncthreads();
    for (int off = 128; off; off >>= 1) {
        if (threadIdx.x < off) red[threadIdx.x] += red[threadIdx.x + off];
        __syncthreads();
    }
    if (threadIdx.x == 0) out[b] = red[0] + bd2[0] + 0.5f;
}

// ---------------------------------------------------------------------------
// Host launcher
// ---------------------------------------------------------------------------
static inline size_t alignUp(size_t v, size_t a) { return (v + a - 1) & ~(a - 1); }

extern "C" void kernel_launch(void* const* d_in, const int* in_sizes, int n_in,
                              void* d_out, int out_size, void* d_ws, size_t ws_size,
                              hipStream_t stream) {
    (void)in_sizes; (void)n_in; (void)out_size; (void)ws_size;
    const float* x  = (const float*)d_in[0];
    const int*   ei = (const int*)d_in[1];
    const float* Wg[4]  = {(const float*)d_in[3],  (const float*)d_in[7],
                           (const float*)d_in[11], (const float*)d_in[15]};
    const float* As[4]  = {(const float*)d_in[4],  (const float*)d_in[8],
                           (const float*)d_in[12], (const float*)d_in[16]};
    const float* Ad[4]  = {(const float*)d_in[5],  (const float*)d_in[9],
                           (const float*)d_in[13], (const float*)d_in[17]};
    const float* Bg[4]  = {(const float*)d_in[6],  (const float*)d_in[10],
                           (const float*)d_in[14], (const float*)d_in[18]};
    const float* Wd1 = (const float*)d_in[19];
    const float* bd1 = (const float*)d_in[20];
    const float* gln = (const float*)d_in[21];
    const float* bln = (const float*)d_in[22];
    const float* Wd2 = (const float*)d_in[23];
    const float* bd2 = (const float*)d_in[24];

    // --- workspace carve ---
    char* ws = (char*)d_ws; size_t off = 0;
    auto carve = [&](size_t bytes) { void* p = ws + off; off = alignUp(off + bytes, 256); return p; };
    float* cbuf            = (float*)carve((size_t)NN * DCAT * 4);        // concat of c1..c4
    __hip_bfloat16* xb     = (__hip_bfloat16*)carve((size_t)NN * 1024 * 2);
    __hip_bfloat16* bp     = (__hip_bfloat16*)carve((size_t)DCAT * HID * 2);
    float* hbuf            = (float*)carve((size_t)NN * 1024 * 4);
    float* ssrc            = (float*)carve((size_t)NN * HEADS * 4);
    float* sdst            = (float*)carve((size_t)NN * HEADS * 4);
    float* mbuf            = (float*)carve((size_t)NN * HEADS * 4);
    float* dbuf            = (float*)carve((size_t)NN * HEADS * 4);
    float* ewbuf           = (float*)carve((size_t)ETOT * HEADS * 4);
    float* pooled          = (float*)carve((size_t)BATCH * DCAT * 4);
    __hip_bfloat16* pooledb= (__hip_bfloat16*)carve((size_t)BATCH * DCAT * 2);
    float* h1              = (float*)carve((size_t)BATCH * HID * 4);

    const int cin[4]    = {768, 512, 512, 1024};
    const int Cc[4]     = {128, 128, 256, 256};
    const int coutT[4]  = {512, 512, 1024, 1024};
    const int colOff[4] = {0, 512, 1024, 2048};

    for (int l = 0; l < 4; ++l) {
        // 1) input -> bf16
        size_t tconv = (size_t)NN * cin[l];
        if (l == 0)
            conv_bf16_kernel<<<(unsigned)((tconv + 255) / 256), 256, 0, stream>>>(
                x, 768, 768, xb, tconv);
        else
            conv_bf16_kernel<<<(unsigned)((tconv + 255) / 256), 256, 0, stream>>>(
                cbuf + colOff[l - 1], DCAT, coutT[l - 1], xb, tconv);
        // 2) pack weights
        size_t tw = (size_t)cin[l] * coutT[l];
        pack_B_kernel<<<(unsigned)((tw + 255) / 256), 256, 0, stream>>>(
            Wg[l], bp, cin[l], coutT[l]);
        // 3) h = x @ W (bf16 WMMA, fp32 accumulate; async/TDM staged)
        dim3 gg(coutT[l] / 64, (NN + 127) / 128);
        wmma_gemm_bf16<<<gg, 256, 0, stream>>>(xb, bp, hbuf, nullptr, NN, cin[l], coutT[l]);
        // 4) per-node attention scores
        int nwaves = NN * HEADS;
        node_scores_kernel<<<(nwaves + 3) / 4, 128, 0, stream>>>(
            hbuf, As[l], Ad[l], ssrc, sdst, NN, HEADS, Cc[l]);
        // 5) init output tile (bias) + softmax state
        size_t tinit = (size_t)NN * coutT[l];
        init_layer_kernel<<<(unsigned)((tinit + 255) / 256), 256, 0, stream>>>(
            cbuf + colOff[l], Bg[l], DCAT, coutT[l], NN, mbuf, dbuf, NN * HEADS);
        // 6-8) edge passes
        int teh = ETOT * HEADS;
        edge_logits_kernel<<<(teh + 255) / 256, 256, 0, stream>>>(
            ei, ssrc, sdst, ewbuf, mbuf, EE, ETOT, HEADS);
        edge_exp_kernel<<<(teh + 255) / 256, 256, 0, stream>>>(
            ei, ewbuf, mbuf, dbuf, EE, ETOT, HEADS);
        edge_agg_kernel<<<ETOT, 256, 0, stream>>>(
            ei, hbuf, ewbuf, dbuf, cbuf + colOff[l], EE, HEADS, Cc[l], DCAT);
    }

    // pooling (ReLU folded in)
    pool_mean_relu_kernel<<<dim3((DCAT + 255) / 256, BATCH), 256, 0, stream>>>(cbuf, pooled);

    // dense head GEMM via WMMA
    size_t tp = (size_t)BATCH * DCAT;
    conv_bf16_kernel<<<(unsigned)((tp + 255) / 256), 256, 0, stream>>>(
        pooled, DCAT, DCAT, pooledb, tp);
    size_t twd = (size_t)DCAT * HID;
    pack_B_kernel<<<(unsigned)((twd + 255) / 256), 256, 0, stream>>>(Wd1, bp, DCAT, HID);
    dim3 gh(HID / 64, (BATCH + 127) / 128);
    wmma_gemm_bf16<<<gh, 256, 0, stream>>>(pooledb, bp, h1, bd1, BATCH, DCAT, HID);

    relu_ln_kernel<<<BATCH, 256, 0, stream>>>(h1, gln, bln, HID);
    final_dot_kernel<<<BATCH, 256, 0, stream>>>(h1, Wd2, bd2, (float*)d_out, HID);
}
